// PointPatchTransformer_23459111371272
// MI455X (gfx1250) — matrix operand order repaired and checked by reading, not compiled, not run on hardware
//
#include <hip/hip_runtime.h>
#include <math.h>

// ---------------------------------------------------------------------------
// Point Patch Transformer for MI455X (gfx1250), wave32 + WMMA f16 (f32 accum)
// ---------------------------------------------------------------------------
#define BB    2
#define NN    10000
#define SS    512
#define KSAMP 64
#define NTOK  513
#define NPAD  528      // 513 padded to multiple of 16 for WMMA M tiles
#define DIM   512
#define INNER 512
#define MLP   1024
#define DEPTH 6
#define R2    0.04f    // RADIUS^2
#define LNEPS 1e-5f

typedef __attribute__((ext_vector_type(16))) _Float16 v16h;
typedef __attribute__((ext_vector_type(8)))  _Float16 v8h;
typedef __attribute__((ext_vector_type(8)))  float    v8f;

// A-fragment (16x32 f16): lane l holds row m=l%16; with h=l/16 this code loads (contiguous k order)
// X[m][k+16h .. +7] into elems 0..7 and X[m][k+16h+8 .. +7] into elems 8..15.
__device__ __forceinline__ v16h frag_a(const _Float16* p0, const _Float16* p1) {
  v8h a0 = *(const v8h*)p0;
  v8h a1 = *(const v8h*)p1;
  return __builtin_shufflevector(a0, a1, 0,1,2,3,4,5,6,7,8,9,10,11,12,13,14,15);
}
// B-fragment (32x16 f16, B[k][n] = W[n][k]): lane l (col n=l%16) loads the 16
// contiguous halves W[n][k+16h .. k+16h+15].
__device__ __forceinline__ v8f wmma_f16(v16h a, v16h b, v8f c) {
  return __builtin_amdgcn_wmma_f32_16x16x32_f16(false, a, false, b, (short)0, c,
                                                false, false);
}

// ---------------------------------------------------------------------------
// Generic GEMM: Out[m,n] = act( sum_k A[m,k]*W[n,k] + bias[n] + resid[m,n] )
// A: f16 [M,K] row-major, W: f16 [N,K] row-major, Out/resid: f32 [M,N].
// Block = 4 waves; each wave owns a 16x64 strip (4 N-tiles) so one A-fragment
// feeds 4 WMMAs. All four B-fragments are loaded into distinct registers
// BEFORE the WMMA burst so the compiler can use partial s_wait_loadcnt and
// issue the WMMAs back-to-back (loads for the next K-step overlap the burst).
// act: 0 none, 2 exact gelu. N must be a multiple of 256, K of 32, M of 16.
// ---------------------------------------------------------------------------
__global__ void __launch_bounds__(128) gemm_wmma(
    const _Float16* __restrict__ A, const _Float16* __restrict__ W,
    const float* __restrict__ bias, const float* __restrict__ resid,
    float* __restrict__ Out, int M, int N, int K, int act)
{
  int lane = threadIdx.x & 31, wv = threadIdx.x >> 5;
  int hl = lane >> 4, l16 = lane & 15;
  int tn0 = (blockIdx.x * 4 + wv) * 4;    // first of 4 consecutive 16-col tiles
  int tm  = blockIdx.y;
  const _Float16* ap = A + (size_t)(tm * 16 + l16) * K + 16 * hl;
  const _Float16* bp0 = W + (size_t)((tn0 + 0) * 16 + l16) * K + 16 * hl;
  const _Float16* bp1 = W + (size_t)((tn0 + 1) * 16 + l16) * K + 16 * hl;
  const _Float16* bp2 = W + (size_t)((tn0 + 2) * 16 + l16) * K + 16 * hl;
  const _Float16* bp3 = W + (size_t)((tn0 + 3) * 16 + l16) * K + 16 * hl;
  v8f c0 = {0.f,0.f,0.f,0.f,0.f,0.f,0.f,0.f};
  v8f c1 = c0, c2 = c0, c3 = c0;
  for (int k = 0; k < K; k += 32) {
    __builtin_prefetch((const void*)(bp0 + k + 128), 0, 1); // global_prefetch
    __builtin_prefetch((const void*)(bp1 + k + 128), 0, 1);
    __builtin_prefetch((const void*)(bp2 + k + 128), 0, 1);
    __builtin_prefetch((const void*)(bp3 + k + 128), 0, 1);
    v16h a  = frag_a(ap + k, ap + k + 8);
    v16h b0 = *(const v16h*)(bp0 + k);
    v16h b1 = *(const v16h*)(bp1 + k);
    v16h b2 = *(const v16h*)(bp2 + k);
    v16h b3 = *(const v16h*)(bp3 + k);
    c0 = wmma_f16(a, b0, c0);
    c1 = wmma_f16(a, b1, c1);
    c2 = wmma_f16(a, b2, c2);
    c3 = wmma_f16(a, b3, c3);
  }
  int rbase = tm * 16 + 8 * hl;     // C/D layout: vgpr r -> row r + 8*(lane/16)
  v8f acc[4] = {c0, c1, c2, c3};
#pragma unroll
  for (int t = 0; t < 4; ++t) {
    int col = (tn0 + t) * 16 + l16;
    float bs = bias ? bias[col] : 0.f;
    for (int r = 0; r < 8; ++r) {
      float v = acc[t][r] + bs;
      size_t off = (size_t)(rbase + r) * N + col;
      if (resid) v += resid[off];
      if (act == 2) v = 0.5f * v * (1.f + erff(v * 0.70710678118f)); // exact gelu
      Out[off] = v;
    }
  }
}

// ---------------------------------------------------------------------------
// Weight casts: f32 -> f16 (transposed or K-padded) done once per launch.
// ---------------------------------------------------------------------------
__global__ void cvt_w_transpose(const float* __restrict__ src,  // [L][K][N]
                                _Float16* __restrict__ dst,     // [L][N][K]
                                int K, int N, long total)
{
  long idx = (long)blockIdx.x * blockDim.x + threadIdx.x;
  if (idx >= total) return;
  long per = (long)K * N;
  long l = idx / per, rem = idx - l * per;
  int n = (int)(rem / K), k = (int)(rem % K);
  dst[idx] = (_Float16)src[l * per + (long)k * N + n];
}
__global__ void cvt_w_pad(const float* __restrict__ src,        // [N][K]
                          _Float16* __restrict__ dst,           // [N][Kp]
                          int N, int K, int Kp)
{
  int idx = blockIdx.x * blockDim.x + threadIdx.x;
  if (idx >= N * Kp) return;
  int n = idx / Kp, k = idx % Kp;
  dst[idx] = (_Float16)((k < K) ? src[n * K + k] : 0.f);
}
// Activation cast [B][NPAD][W] f32 -> f16, zeroing pad rows (r >= valid).
__global__ void cvt_a16(const float* __restrict__ in, _Float16* __restrict__ out,
                        int W, int valid)
{
  size_t idx = (size_t)blockIdx.x * blockDim.x + threadIdx.x;
  size_t total = (size_t)BB * NPAD * W;
  if (idx >= total) return;
  int r = (int)((idx / W) % NPAD);
  float v = (r < valid) ? in[idx] : 0.f;
  out[idx] = (_Float16)v;
}

// ---------------------------------------------------------------------------
// Geometry: transpose, FPS (LDS-resident distance array), ball query + group
// ---------------------------------------------------------------------------
__global__ void transpose_xyz(const float* __restrict__ f, float* __restrict__ xyz)
{
  int idx = blockIdx.x * blockDim.x + threadIdx.x;
  if (idx >= BB * NN) return;
  int b = idx / NN, j = idx % NN;
  for (int d = 0; d < 3; ++d)
    xyz[(size_t)idx * 3 + d] = f[(size_t)b * 3 * NN + (size_t)d * NN + j];
}

__global__ void __launch_bounds__(256) fps_kernel(const float* __restrict__ xyz,
                                                  float* __restrict__ new_xyz)
{
  __shared__ float sd[NN];          // 40 KB of the 320 KB WGP LDS pool
  __shared__ float rv[256];
  __shared__ int   ri[256];
  __shared__ float cen[3];
  __shared__ int   sfar;
  int b = blockIdx.x, tid = threadIdx.x;
  const float* base = xyz + (size_t)b * NN * 3;
  for (int j = tid; j < NN; j += 256) sd[j] = 1e10f;
  if (tid == 0) sfar = 0;
  __syncthreads();
  for (int it = 0; it < SS; ++it) {
    int f = sfar;
    if (tid < 3) {
      float c = base[f * 3 + tid];
      cen[tid] = c;
      new_xyz[((size_t)b * SS + it) * 3 + tid] = c;
    }
    __syncthreads();
    float cx = cen[0], cy = cen[1], cz = cen[2];
    float best = -1.f; int bi = 0;
    for (int j = tid; j < NN; j += 256) {
      float dx = base[j*3] - cx, dy = base[j*3+1] - cy, dz = base[j*3+2] - cz;
      float d = dx*dx + dy*dy + dz*dz;
      float nd = fminf(sd[j], d); sd[j] = nd;
      if (nd > best) { best = nd; bi = j; }
    }
    rv[tid] = best; ri[tid] = bi;
    __syncthreads();
    for (int s = 128; s > 0; s >>= 1) {
      if (tid < s) {
        float ov = rv[tid + s]; int oi = ri[tid + s];
        if (ov > rv[tid] || (ov == rv[tid] && oi < ri[tid])) { rv[tid]=ov; ri[tid]=oi; }
      }
      __syncthreads();
    }
    if (tid == 0) sfar = ri[0];
    __syncthreads();
  }
}

// Ball query (first 64 indices in order == smallest valid indices) + build
// grouped f16 input [g][64][32]: cols 0..2 = p - q, 3..5 = p, 6..31 = 0.
__global__ void ball_group(const float* __restrict__ xyz,
                           const float* __restrict__ new_xyz,
                           _Float16* __restrict__ groups,
                           float* __restrict__ cent)   // [B][3][513]
{
  int id = blockIdx.x * blockDim.x + threadIdx.x;
  if (id >= BB * SS) return;
  int b = id >> 9, s = id & 511;
  const float* base = xyz + (size_t)b * NN * 3;
  float qx = new_xyz[id*3], qy = new_xyz[id*3+1], qz = new_xyz[id*3+2];
  int idxs[KSAMP]; int cnt = 0;
  for (int j = 0; j < NN; ++j) {
    float dx = base[j*3]-qx, dy = base[j*3+1]-qy, dz = base[j*3+2]-qz;
    if (dx*dx + dy*dy + dz*dz <= R2) { idxs[cnt++] = j; if (cnt == KSAMP) break; }
  }
  if (cnt == 0) { idxs[0] = 0; cnt = 1; }   // cannot happen (self in ball)
  for (int k = cnt; k < KSAMP; ++k) idxs[k] = idxs[0];
  _Float16* gp = groups + (size_t)id * KSAMP * 32;
  for (int k = 0; k < KSAMP; ++k) {
    int j = idxs[k];
    float px = base[j*3], py = base[j*3+1], pz = base[j*3+2];
    _Float16* r = gp + k * 32;
    r[0]=(_Float16)(px-qx); r[1]=(_Float16)(py-qy); r[2]=(_Float16)(pz-qz);
    r[3]=(_Float16)px;      r[4]=(_Float16)py;      r[5]=(_Float16)pz;
    for (int c = 6; c < 32; ++c) r[c] = (_Float16)0.f;
  }
  float* cb = cent + (size_t)b * 3 * NTOK;
  cb[0*NTOK + 1 + s] = qx; cb[1*NTOK + 1 + s] = qy; cb[2*NTOK + 1 + s] = qz;
  if (s == 0) { cb[0] = 0.f; cb[NTOK] = 0.f; cb[2*NTOK] = 0.f; }
}

// ---------------------------------------------------------------------------
// Set-abstraction MLP (6->64->64->256, BN folded, relu) + maxpool over 64
// samples, all via WMMA out of LDS. One workgroup (4 waves) per group.
// A-fragments hoisted over N tiles; B fragments preloaded before WMMA bursts.
// Writes the lift input row [288] = [cent(3) | feat(256) | zeros(29)] as f16.
// ---------------------------------------------------------------------------
__global__ void __launch_bounds__(128) sa_mlp_kernel(
    const _Float16* __restrict__ groups,
    const _Float16* __restrict__ w0, const _Float16* __restrict__ w1,
    const _Float16* __restrict__ w2,
    const float* __restrict__ cb0, const float* __restrict__ g0,
    const float* __restrict__ be0, const float* __restrict__ m0,
    const float* __restrict__ v0,
    const float* __restrict__ cb1, const float* __restrict__ g1,
    const float* __restrict__ be1, const float* __restrict__ m1,
    const float* __restrict__ v1,
    const float* __restrict__ cb2, const float* __restrict__ g2,
    const float* __restrict__ be2, const float* __restrict__ m2,
    const float* __restrict__ v2,
    const float* __restrict__ new_xyz,
    _Float16* __restrict__ liftin)
{
  __shared__ __align__(16) _Float16 sX[64 * 32];
  __shared__ __align__(16) _Float16 sY[64 * 64];
  __shared__ __align__(16) _Float16 sZ[64 * 64];
  __shared__ float pm[8 * 256];
  int g = blockIdx.x, tid = threadIdx.x;
  { // stage grouped input
    const float4* src = (const float4*)(groups + (size_t)g * 2048);
    float4* dst = (float4*)sX;
    for (int i = tid; i < 256; i += 128) dst[i] = src[i];
  }
  __syncthreads();
  int lane = tid & 31, wv = tid >> 5, hl = lane >> 4, l16 = lane & 15;
  int row  = wv * 16 + l16;        // A row for this wave's M tile
  int orow = wv * 16 + 8 * hl;     // C/D row base
  // layer 0: [64,32] x [64,32]^T  (K padded 6->32): 1 A frag, 4 WMMAs
  {
    v16h a  = frag_a(sX + row*32 + 16*hl, sX + row*32 + 8 + 16*hl);
    v16h b0 = *(const v16h*)(w0 + ( 0 + l16)*32 + 16*hl);
    v16h b1 = *(const v16h*)(w0 + (16 + l16)*32 + 16*hl);
    v16h b2 = *(const v16h*)(w0 + (32 + l16)*32 + 16*hl);
    v16h b3 = *(const v16h*)(w0 + (48 + l16)*32 + 16*hl);
    v8f z = {0.f,0.f,0.f,0.f,0.f,0.f,0.f,0.f};
    v8f c[4];
    c[0] = wmma_f16(a, b0, z);
    c[1] = wmma_f16(a, b1, z);
    c[2] = wmma_f16(a, b2, z);
    c[3] = wmma_f16(a, b3, z);
#pragma unroll
    for (int t = 0; t < 4; ++t) {
      int col = 16*t + l16;
      float sc = g0[col] * rsqrtf(v0[col] + LNEPS);
      float sh = (cb0[col] - m0[col]) * sc + be0[col];
      for (int r = 0; r < 8; ++r)
        sY[(orow + r)*64 + col] = (_Float16)fmaxf(c[t][r]*sc + sh, 0.f);
    }
  }
  __syncthreads();
  // layer 1: [64,64] x [64,64]^T, A hoisted, B preloaded
  {
    v8f c[4];
#pragma unroll
    for (int t = 0; t < 4; ++t) c[t] = (v8f){0.f,0.f,0.f,0.f,0.f,0.f,0.f,0.f};
    for (int ks = 0; ks < 64; ks += 32) {
      v16h a  = frag_a(sY + row*64 + ks + 16*hl, sY + row*64 + ks + 8 + 16*hl);
      v16h b0 = *(const v16h*)(w1 + ( 0 + l16)*64 + ks + 16*hl);
      v16h b1 = *(const v16h*)(w1 + (16 + l16)*64 + ks + 16*hl);
      v16h b2 = *(const v16h*)(w1 + (32 + l16)*64 + ks + 16*hl);
      v16h b3 = *(const v16h*)(w1 + (48 + l16)*64 + ks + 16*hl);
      c[0] = wmma_f16(a, b0, c[0]);
      c[1] = wmma_f16(a, b1, c[1]);
      c[2] = wmma_f16(a, b2, c[2]);
      c[3] = wmma_f16(a, b3, c[3]);
    }
#pragma unroll
    for (int t = 0; t < 4; ++t) {
      int col = 16*t + l16;
      float sc = g1[col] * rsqrtf(v1[col] + LNEPS);
      float sh = (cb1[col] - m1[col]) * sc + be1[col];
      for (int r = 0; r < 8; ++r)
        sZ[(orow + r)*64 + col] = (_Float16)fmaxf(c[t][r]*sc + sh, 0.f);
    }
  }
  __syncthreads();
  // layer 2: [64,64] x [256,64]^T in 4 groups of 4 N-tiles, fused row-max
  for (int grp = 0; grp < 4; ++grp) {
    const _Float16* wg = w2 + (size_t)(64 * grp) * 64;
    v8f c[4];
#pragma unroll
    for (int t = 0; t < 4; ++t) c[t] = (v8f){0.f,0.f,0.f,0.f,0.f,0.f,0.f,0.f};
    for (int ks = 0; ks < 64; ks += 32) {
      v16h a  = frag_a(sZ + row*64 + ks + 16*hl, sZ + row*64 + ks + 8 + 16*hl);
      v16h b0 = *(const v16h*)(wg + ( 0 + l16)*64 + ks + 16*hl);
      v16h b1 = *(const v16h*)(wg + (16 + l16)*64 + ks + 16*hl);
      v16h b2 = *(const v16h*)(wg + (32 + l16)*64 + ks + 16*hl);
      v16h b3 = *(const v16h*)(wg + (48 + l16)*64 + ks + 16*hl);
      c[0] = wmma_f16(a, b0, c[0]);
      c[1] = wmma_f16(a, b1, c[1]);
      c[2] = wmma_f16(a, b2, c[2]);
      c[3] = wmma_f16(a, b3, c[3]);
    }
#pragma unroll
    for (int t = 0; t < 4; ++t) {
      int col = 16*(grp*4 + t) + l16;
      float sc = g2[col] * rsqrtf(v2[col] + LNEPS);
      float sh = (cb2[col] - m2[col]) * sc + be2[col];
      float mx = -1e30f;
      for (int r = 0; r < 8; ++r) mx = fmaxf(mx, fmaxf(c[t][r]*sc + sh, 0.f));
      pm[(wv*2 + hl)*256 + col] = mx;
    }
  }
  __syncthreads();
  size_t lb = (size_t)g * 288;
  for (int ch = tid; ch < 256; ch += 128) {
    float mx = pm[ch];
    for (int q = 1; q < 8; ++q) mx = fmaxf(mx, pm[q*256 + ch]);
    liftin[lb + 3 + ch] = (_Float16)mx;
  }
  if (tid < 3)              liftin[lb + tid]       = (_Float16)new_xyz[(size_t)g*3 + tid];
  if (tid >= 3 && tid < 32) liftin[lb + 256 + tid] = (_Float16)0.f;   // cols 259..287
}

// ---------------------------------------------------------------------------
// LayerNorm kernels (block-per-row, fp32)
// ---------------------------------------------------------------------------
__global__ void __launch_bounds__(256) lift_ln(const float* __restrict__ in,
                                               const float* __restrict__ g,
                                               const float* __restrict__ be,
                                               float* __restrict__ x)
{
  int blk = blockIdx.x;                 // BB*SS rows of liftout
  int b = blk >> 9, s = blk & 511;
  const float* row = in + (size_t)blk * DIM;
  float* orow = x + ((size_t)b * NPAD + 1 + s) * DIM;
  __shared__ float r1[256], r2[256];
  int tid = threadIdx.x;
  float sm = 0.f, sq = 0.f;
  for (int c = tid; c < DIM; c += 256) { float v = row[c]; sm += v; sq += v*v; }
  r1[tid] = sm; r2[tid] = sq; __syncthreads();
  for (int st = 128; st; st >>= 1) {
    if (tid < st) { r1[tid] += r1[tid+st]; r2[tid] += r2[tid+st]; }
    __syncthreads();
  }
  float m = r1[0] * (1.f/DIM);
  float rs = rsqrtf(r2[0]*(1.f/DIM) - m*m + LNEPS);
  for (int c = tid; c < DIM; c += 256) orow[c] = (row[c]-m)*rs*g[c] + be[c];
}

__global__ void __launch_bounds__(256) ln_rows(const float* __restrict__ in,
                                               float* __restrict__ out,
                                               const float* __restrict__ g,
                                               const float* __restrict__ be)
{
  int blk = blockIdx.x;                 // BB*NTOK
  int b = blk / NTOK, r = blk % NTOK;
  size_t row = ((size_t)b * NPAD + r) * DIM;
  __shared__ float r1[256], r2[256];
  int tid = threadIdx.x;
  float sm = 0.f, sq = 0.f;
  for (int c = tid; c < DIM; c += 256) { float v = in[row+c]; sm += v; sq += v*v; }
  r1[tid] = sm; r2[tid] = sq; __syncthreads();
  for (int st = 128; st; st >>= 1) {
    if (tid < st) { r1[tid] += r1[tid+st]; r2[tid] += r2[tid+st]; }
    __syncthreads();
  }
  float m = r1[0] * (1.f/DIM);
  float rs = rsqrtf(r2[0]*(1.f/DIM) - m*m + LNEPS);
  for (int c = tid; c < DIM; c += 256) out[row+c] = (in[row+c]-m)*rs*g[c] + be[c];
}

// ---------------------------------------------------------------------------
// Pairwise positional bias: pe[b,i,j] = b2 + sum_c w2[c]*relu(w1[c]·(ci-cj)+b1[c])
// ---------------------------------------------------------------------------
__global__ void __launch_bounds__(256) pe_map(const float* __restrict__ cent,
                                              const float* __restrict__ w1,
                                              const float* __restrict__ b1,
                                              const float* __restrict__ w2,
                                              const float* __restrict__ b2,
                                              float* __restrict__ pe)
{
  int blk = blockIdx.x;                 // BB*NTOK
  int i = blk % NTOK, b = blk / NTOK;
  __shared__ float sw1[192], sb1[64], sw2[64];
  int tid = threadIdx.x;
  if (tid < 192) sw1[tid] = w1[tid];
  if (tid < 64) { sb1[tid] = b1[tid]; sw2[tid] = w2[tid]; }
  __syncthreads();
  const float* cb = cent + (size_t)b * 3 * NTOK;
  float cx = cb[i], cy = cb[NTOK + i], cz = cb[2*NTOK + i];
  float bb2 = b2[0];
  for (int j = tid; j < NTOK; j += 256) {
    float dx = cx - cb[j], dy = cy - cb[NTOK + j], dz = cz - cb[2*NTOK + j];
    float acc = bb2;
    for (int c = 0; c < 64; ++c) {
      float t = sw1[c*3]*dx + sw1[c*3+1]*dy + sw1[c*3+2]*dz + sb1[c];
      acc += sw2[c] * fmaxf(t, 0.f);
    }
    pe[((size_t)b * NTOK + i) * NTOK + j] = acc;
  }
}

// ---------------------------------------------------------------------------
// Attention: one block per (b, h, query i). dots->softmax->o, fp32.
// qkv layout [B][NPAD][1536] with q|k|v thirds, head h = 64-col slice.
// ---------------------------------------------------------------------------
__global__ void __launch_bounds__(256) attn_kernel(const float* __restrict__ qkv,
                                                   const float* __restrict__ pe,
                                                   float* __restrict__ o)
{
  int blk = blockIdx.x;
  int i = blk % NTOK, h = (blk / NTOK) & 7, b = blk / (NTOK * 8);
  __shared__ float sq[64], sd[NTOK], red[256], oacc[4][64];
  int tid = threadIdx.x;
  const float* qrow = qkv + ((size_t)(b*NPAD + i) * 1536) + h*64;
  if (tid < 64) sq[tid] = qrow[tid];
  __syncthreads();
  const size_t kbase = (size_t)b * NPAD * 1536 + 512 + h*64;
  const float* perow = pe + ((size_t)b * NTOK + i) * NTOK;
  const float scale = 0.125f;           // DH^-0.5
  for (int j = tid; j < NTOK; j += 256) {
    const float* krow = qkv + kbase + (size_t)j * 1536;
    float acc = 0.f;
    for (int d = 0; d < 64; ++d) acc += sq[d] * krow[d];
    sd[j] = (acc + perow[j]) * scale;
  }
  __syncthreads();
  float mx = -1e30f;
  for (int j = tid; j < NTOK; j += 256) mx = fmaxf(mx, sd[j]);
  red[tid] = mx; __syncthreads();
  for (int st = 128; st; st >>= 1) {
    if (tid < st) red[tid] = fmaxf(red[tid], red[tid+st]);
    __syncthreads();
  }
  float gmax = red[0]; __syncthreads();
  float ssum = 0.f;
  for (int j = tid; j < NTOK; j += 256) { float e = __expf(sd[j]-gmax); sd[j]=e; ssum+=e; }
  red[tid] = ssum; __syncthreads();
  for (int st = 128; st; st >>= 1) {
    if (tid < st) red[tid] += red[tid+st];
    __syncthreads();
  }
  float inv = 1.f / red[0]; __syncthreads();
  int part = tid >> 6, d = tid & 63;
  const size_t vbase = (size_t)b * NPAD * 1536 + 1024 + h*64 + d;
  float acc = 0.f;
  for (int j = part; j < NTOK; j += 4) acc += sd[j] * qkv[vbase + (size_t)j * 1536];
  oacc[part][d] = acc;
  __syncthreads();
  if (part == 0)
    o[((size_t)(b*NPAD + i)) * 512 + h*64 + d] =
        (oacc[0][d] + oacc[1][d] + oacc[2][d] + oacc[3][d]) * inv;
}

// ---------------------------------------------------------------------------
// Misc small kernels
// ---------------------------------------------------------------------------
__global__ void init_x(const float* __restrict__ cls, float* __restrict__ x)
{
  int idx = blockIdx.x * blockDim.x + threadIdx.x;
  if (idx >= BB * 16 * DIM) return;
  int b = idx / (16 * DIM), rr = (idx / DIM) % 16, c = idx % DIM;
  if (rr == 0) x[(size_t)b * NPAD * DIM + c] = cls[c];
  else x[((size_t)b * NPAD + 512 + rr) * DIM + c] = 0.f;   // pad rows 513..527
}
__global__ void extract_out(const float* __restrict__ x, float* __restrict__ out)
{
  int idx = blockIdx.x * blockDim.x + threadIdx.x;
  if (idx >= BB * DIM) return;
  out[idx] = x[(size_t)(idx >> 9) * NPAD * DIM + (idx & 511)];
}

// ---------------------------------------------------------------------------
// Host orchestration (all launches on `stream`; no allocation/sync -> graph safe)
// ---------------------------------------------------------------------------
static inline void launch_gemm(const _Float16* A, const _Float16* W,
                               const float* bias, const float* resid, float* Out,
                               int M, int N, int K, int act, hipStream_t s)
{
  dim3 grid(N / 256, M / 16);   // 4 waves x 64 cols each
  gemm_wmma<<<grid, 128, 0, s>>>(A, W, bias, resid, Out, M, N, K, act);
}

extern "C" void kernel_launch(void* const* d_in, const int* in_sizes, int n_in,
                              void* d_out, int out_size, void* d_ws, size_t ws_size,
                              hipStream_t stream)
{
  (void)in_sizes; (void)n_in; (void)out_size; (void)ws_size;
  const float* features  = (const float*)d_in[0];
  const float* sa_w0     = (const float*)d_in[1];
  const float* sa_b0     = (const float*)d_in[2];
  const float* sa_w1     = (const float*)d_in[3];
  const float* sa_b1     = (const float*)d_in[4];
  const float* sa_w2     = (const float*)d_in[5];
  const float* sa_b2     = (const float*)d_in[6];
  const float* bn_g0 = (const float*)d_in[7];  const float* bn_b0 = (const float*)d_in[8];
  const float* bn_m0 = (const float*)d_in[9];  const float* bn_v0 = (const float*)d_in[10];
  const float* bn_g1 = (const float*)d_in[11]; const float* bn_b1 = (const float*)d_in[12];
  const float* bn_m1 = (const float*)d_in[13]; const float* bn_v1 = (const float*)d_in[14];
  const float* bn_g2 = (const float*)d_in[15]; const float* bn_b2 = (const float*)d_in[16];
  const float* bn_m2 = (const float*)d_in[17]; const float* bn_v2 = (const float*)d_in[18];
  const float* lift_w    = (const float*)d_in[19];
  const float* lift_b    = (const float*)d_in[20];
  const float* lift_g    = (const float*)d_in[21];
  const float* lift_beta = (const float*)d_in[22];
  const float* cls_token = (const float*)d_in[23];
  const float* qkv_w = (const float*)d_in[24];
  const float* out_w = (const float*)d_in[25];
  const float* out_b = (const float*)d_in[26];
  const float* ln1_g = (const float*)d_in[27];
  const float* ln1_b = (const float*)d_in[28];
  const float* pe_w1 = (const float*)d_in[29];
  const float* pe_b1 = (const float*)d_in[30];
  const float* pe_w2 = (const float*)d_in[31];
  const float* pe_b2 = (const float*)d_in[32];
  const float* ln2_g = (const float*)d_in[33];
  const float* ln2_b = (const float*)d_in[34];
  const float* ff_w1 = (const float*)d_in[35];
  const float* ff_b1 = (const float*)d_in[36];
  const float* ff_w2 = (const float*)d_in[37];
  const float* ff_b2 = (const float*)d_in[38];

  // ---- workspace carve (256B aligned) ----
  char* wsb = (char*)d_ws; size_t off = 0;
  auto alloc = [&](size_t bytes) -> void* {
    void* p = wsb + off; off += (bytes + 255) & ~(size_t)255; return p;
  };
  float*    xyz     = (float*)alloc((size_t)BB*NN*3*4);
  float*    nxyz    = (float*)alloc((size_t)BB*SS*3*4);
  float*    cent    = (float*)alloc((size_t)BB*3*NTOK*4);
  _Float16* groups  = (_Float16*)alloc((size_t)BB*SS*KSAMP*32*2);
  _Float16* liftin  = (_Float16*)alloc((size_t)BB*SS*288*2);
  float*    liftout = (float*)alloc((size_t)BB*SS*DIM*4);
  float*    x       = (float*)alloc((size_t)BB*NPAD*DIM*4);
  float*    tmpA    = (float*)alloc((size_t)BB*NPAD*1536*4);   // qkv
  float*    tmpB    = (float*)alloc((size_t)BB*NPAD*1024*4);   // h1/o/h2/g
  _Float16* a16     = (_Float16*)alloc((size_t)BB*NPAD*1024*2);
  float*    peb     = (float*)alloc((size_t)BB*NTOK*NTOK*4);
  _Float16* qkvw16  = (_Float16*)alloc((size_t)DEPTH*1536*512*2);
  _Float16* outw16  = (_Float16*)alloc((size_t)DEPTH*512*512*2);
  _Float16* f1w16   = (_Float16*)alloc((size_t)DEPTH*1024*512*2);
  _Float16* f2w16   = (_Float16*)alloc((size_t)DEPTH*512*1024*2);
  _Float16* liftw16 = (_Float16*)alloc((size_t)512*288*2);
  _Float16* saw0    = (_Float16*)alloc((size_t)64*32*2);
  _Float16* saw1    = (_Float16*)alloc((size_t)64*64*2);
  _Float16* saw2    = (_Float16*)alloc((size_t)256*64*2);

  // ---- one-time (per launch) weight casts ----
  { long t = (long)DEPTH*512*1536;
    cvt_w_transpose<<<(int)((t+255)/256),256,0,stream>>>(qkv_w, qkvw16, 512, 1536, t); }
  { long t = (long)DEPTH*512*512;
    cvt_w_transpose<<<(int)((t+255)/256),256,0,stream>>>(out_w, outw16, 512, 512, t); }
  { long t = (long)DEPTH*512*1024;
    cvt_w_transpose<<<(int)((t+255)/256),256,0,stream>>>(ff_w1, f1w16, 512, 1024, t); }
  { long t = (long)DEPTH*1024*512;
    cvt_w_transpose<<<(int)((t+255)/256),256,0,stream>>>(ff_w2, f2w16, 1024, 512, t); }
  cvt_w_pad<<<(512*288+255)/256,256,0,stream>>>(lift_w, liftw16, 512, 259, 288);
  cvt_w_pad<<<(64*32+255)/256,  256,0,stream>>>(sa_w0, saw0, 64, 6, 32);
  cvt_w_pad<<<(64*64+255)/256,  256,0,stream>>>(sa_w1, saw1, 64, 64, 64);
  cvt_w_pad<<<(256*64+255)/256, 256,0,stream>>>(sa_w2, saw2, 256, 64, 64);

  // ---- geometry + set abstraction ----
  transpose_xyz<<<(BB*NN+255)/256,256,0,stream>>>(features, xyz);
  fps_kernel<<<BB,256,0,stream>>>(xyz, nxyz);
  ball_group<<<(BB*SS+127)/128,128,0,stream>>>(xyz, nxyz, groups, cent);
  sa_mlp_kernel<<<BB*SS,128,0,stream>>>(groups, saw0, saw1, saw2,
      sa_b0, bn_g0, bn_b0, bn_m0, bn_v0,
      sa_b1, bn_g1, bn_b1, bn_m1, bn_v1,
      sa_b2, bn_g2, bn_b2, bn_m2, bn_v2,
      nxyz, liftin);

  // ---- lift + LN + token assembly ----
  launch_gemm(liftin, liftw16, lift_b, nullptr, liftout, BB*SS, DIM, 288, 0, stream);
  lift_ln<<<BB*SS,256,0,stream>>>(liftout, lift_g, lift_beta, x);
  init_x<<<(BB*16*DIM+255)/256,256,0,stream>>>(cls_token, x);

  // ---- transformer ----
  const int M = BB * NPAD;
  for (int l = 0; l < DEPTH; ++l) {
    ln_rows<<<BB*NTOK,256,0,stream>>>(x, tmpB, ln1_g + l*DIM, ln1_b + l*DIM);
    { long t = (long)BB*NPAD*DIM;
      cvt_a16<<<(int)((t+255)/256),256,0,stream>>>(tmpB, a16, DIM, NTOK); }
    launch_gemm(a16, qkvw16 + (size_t)l*1536*512, nullptr, nullptr, tmpA,
                M, 1536, 512, 0, stream);
    pe_map<<<BB*NTOK,256,0,stream>>>(cent, pe_w1 + l*192, pe_b1 + l*64,
                                     pe_w2 + l*64, pe_b2 + l, peb);
    attn_kernel<<<BB*8*NTOK,256,0,stream>>>(tmpA, peb, tmpB);
    { long t = (long)BB*NPAD*DIM;
      cvt_a16<<<(int)((t+255)/256),256,0,stream>>>(tmpB, a16, DIM, NTOK); }
    launch_gemm(a16, outw16 + (size_t)l*512*512, out_b + l*DIM, x, x,
                M, 512, 512, 0, stream);                     // x += o@Wo + b
    ln_rows<<<BB*NTOK,256,0,stream>>>(x, tmpB, ln2_g + l*DIM, ln2_b + l*DIM);
    { long t = (long)BB*NPAD*DIM;
      cvt_a16<<<(int)((t+255)/256),256,0,stream>>>(tmpB, a16, DIM, NTOK); }
    launch_gemm(a16, f1w16 + (size_t)l*1024*512, ff_b1 + l*MLP, nullptr, tmpB,
                M, 1024, 512, 2, stream);                    // gelu
    { long t = (long)BB*NPAD*1024;
      cvt_a16<<<(int)((t+255)/256),256,0,stream>>>(tmpB, a16, 1024, NTOK); }
    launch_gemm(a16, f2w16 + (size_t)l*512*1024, ff_b2 + l*DIM, x, x,
                M, 512, 1024, 0, stream);                    // x += ff
  }
  extract_out<<<(BB*DIM+255)/256,256,0,stream>>>(x, (float*)d_out);
}
